// MultiHeadAttention_37589553774750
// MI455X (gfx1250) — matrix operand order repaired and checked by reading, not compiled, not run on hardware
//
#include <hip/hip_runtime.h>
#include <hip/hip_bf16.h>

typedef __bf16 bf16_t;
typedef __attribute__((ext_vector_type(16))) __bf16 v16bf;
typedef __attribute__((ext_vector_type(8)))  __bf16 v8bf;
typedef __attribute__((ext_vector_type(8)))  float  v8f;
typedef __attribute__((ext_vector_type(4)))  float  v4f;
typedef __attribute__((ext_vector_type(4)))  int    v4i;

#define BATCH 4
#define SEQ   2048
#define DIN   1024
#define DOUT  1024
#define NH    16
#define HD    64
#define MTOT  (BATCH * SEQ)

#define WMMA_BF16(A, Bm, C) \
  __builtin_amdgcn_wmma_f32_16x16x32_bf16(false, (A), false, (Bm), (short)0, (C), false, false)

#define AS1 __attribute__((address_space(1)))
#define AS3 __attribute__((address_space(3)))

#if defined(__gfx1250__) && __has_builtin(__builtin_amdgcn_global_load_async_to_lds_b128)
#define HAS_ASYNC_LDS 1
#else
#define HAS_ASYNC_LDS 0
#endif

static __device__ __forceinline__ v16bf cat8(v8bf lo, v8bf hi) {
  return __builtin_shufflevector(lo, hi, 0,1,2,3,4,5,6,7,8,9,10,11,12,13,14,15);
}

// Pack two f32 -> two bf16 in one dword (enables b32/b64 LDS stores).
static __device__ __forceinline__ unsigned pack2bf(float a, float b) {
  unsigned short ua = __builtin_bit_cast(unsigned short, (bf16_t)a);
  unsigned short ub = __builtin_bit_cast(unsigned short, (bf16_t)b);
  return (unsigned)ua | ((unsigned)ub << 16);
}

static __device__ __forceinline__ void async_wait0() {
#if defined(__gfx1250__) && __has_builtin(__builtin_amdgcn_s_wait_asynccnt)
  __builtin_amdgcn_s_wait_asynccnt(0);
#elif defined(__gfx1250__)
  asm volatile("s_wait_asynccnt 0x0" ::: "memory");
#endif
}

// Copy 16 bytes global->LDS: async DMA path (ASYNCcnt) with sync fallback.
static __device__ __forceinline__ void cp16_g2l(const bf16_t* g, bf16_t* l) {
#if HAS_ASYNC_LDS
  __builtin_amdgcn_global_load_async_to_lds_b128(
      (AS1 v4i*)g, (AS3 v4i*)l, 0, 0);
#else
  *(v8bf*)l = *(const v8bf*)g;
#endif
}

// ---------------------------------------------------------------------------
// Kernel 0: one-shot weight convert+transpose. Wt[z][n][k] = bf16(W_z[k][n]).
// grid = (DOUT/32, DIN/32, 4), block = (32, 8). LDS-tiled for coalescing.
// ---------------------------------------------------------------------------
__global__ __launch_bounds__(256)
void wtrans_kernel(const float* __restrict__ Wq, const float* __restrict__ Wk,
                   const float* __restrict__ Wv, const float* __restrict__ Wo,
                   bf16_t* __restrict__ Wt) {
  __shared__ bf16_t tile[32][33];
  const int z = blockIdx.z;
  const float* W = (z == 0) ? Wq : (z == 1) ? Wk : (z == 2) ? Wv : Wo;
  const int n0 = blockIdx.x * 32;
  const int k0 = blockIdx.y * 32;
  const int tx = threadIdx.x, ty = threadIdx.y;
  #pragma unroll
  for (int i = 0; i < 4; ++i) {
    int kk = ty * 4 + i;
    tile[kk][tx] = (bf16_t)W[(size_t)(k0 + kk) * DOUT + n0 + tx];
  }
  __syncthreads();
  bf16_t* dst = Wt + (size_t)z * DIN * DOUT;
  #pragma unroll
  for (int i = 0; i < 4; ++i) {
    int nn = ty * 4 + i;
    dst[(size_t)(n0 + nn) * DIN + k0 + tx] = tile[tx][nn];
  }
}

// ---------------------------------------------------------------------------
// Kernel 1: fused QKV projection. grid = (MTOT/128, DOUT/64, 3), block = 128.
// Block computes a 128x64 tile; each wave owns 32 rows x 64 cols
// (2 A-frags x 4 B-frags -> 8 WMMAs per 32-wide K step, B reused twice).
// X staged f32->bf16 with packed b64 LDS stores; pre-transposed bf16 weights
// staged via async global->LDS b128 DMA.
// Q,K stored [B,H,T,HD]; V stored transposed [B,H,HD,T].
// ---------------------------------------------------------------------------
__global__ __launch_bounds__(128)
void qkv_gemm_kernel(const float* __restrict__ x,
                     const bf16_t* __restrict__ Wt,   // [3][DOUT][DIN] bf16
                     bf16_t* __restrict__ Qb,
                     bf16_t* __restrict__ Kb,
                     bf16_t* __restrict__ Vtb) {
  __shared__ alignas(64) bf16_t Xs[128][32];
  __shared__ alignas(64) bf16_t Ws[64][32];   // [n][k]

  const int m0 = blockIdx.x * 128;
  const int n0 = blockIdx.y * 64;
  const int z  = blockIdx.z;
  const bf16_t* Wz = Wt + (size_t)z * DIN * DOUT;

  const int tid  = threadIdx.x;
  const int wave = tid >> 5;
  const int lane = tid & 31;
  const int row  = lane & 15;
  const int half = lane >> 4;

  const v8f vzero = {0.f,0.f,0.f,0.f,0.f,0.f,0.f,0.f};
  v8f acc[8];
  #pragma unroll
  for (int i = 0; i < 8; ++i) acc[i] = vzero;

  for (int k0 = 0; k0 < DIN; k0 += 32) {
    __syncthreads();
    // Stage X tile: 128 rows x 32 k, f32 -> bf16, packed b64 stores.
    #pragma unroll
    for (int i = 0; i < 8; ++i) {
      int idx = i * 512 + tid * 4;
      int r = idx >> 5, c = idx & 31;
      v4f xv = *(const v4f*)(x + (size_t)(m0 + r) * DIN + k0 + c);
      uint2 u; u.x = pack2bf(xv.x, xv.y); u.y = pack2bf(xv.z, xv.w);
      *(uint2*)&Xs[r][c] = u;
    }
    // Stage Wt tile (64 n x 32 k bf16) via async global->LDS DMA.
    #pragma unroll
    for (int i = 0; i < 2; ++i) {
      int idx = i * 1024 + tid * 8;
      int r = idx >> 5, c = idx & 31;
      cp16_g2l(Wz + (size_t)(n0 + r) * DIN + k0 + c, &Ws[r][c]);
    }
    async_wait0();
    __syncthreads();

    // A fragments (ISA 16-bit 16x32 A layout), 2 row-blocks per wave.
    v16bf a0 = cat8(*(const v8bf*)&Xs[wave * 32 + row][8 * half],
                    *(const v8bf*)&Xs[wave * 32 + row][16 + 8 * half]);
    v16bf a1 = cat8(*(const v8bf*)&Xs[wave * 32 + 16 + row][8 * half],
                    *(const v8bf*)&Xs[wave * 32 + 16 + row][16 + 8 * half]);
    #pragma unroll
    for (int nb = 0; nb < 4; ++nb) {
      v16bf b = cat8(*(const v8bf*)&Ws[nb * 16 + row][8 * half], *(const v8bf*)&Ws[nb * 16 + row][16 + 8 * half]);
      acc[nb]     = WMMA_BF16(a0, b, acc[nb]);
      acc[4 + nb] = WMMA_BF16(a1, b, acc[4 + nb]);
    }
  }

  // Epilogue (z branch hoisted out of the store loops).
  if (z == 2) {
    #pragma unroll
    for (int f = 0; f < 2; ++f)
      #pragma unroll
      for (int nb = 0; nb < 4; ++nb)
        #pragma unroll
        for (int j = 0; j < 8; ++j) {
          int mg  = m0 + wave * 32 + f * 16 + j + 8 * half;
          int col = n0 + nb * 16 + row;
          int bi = mg >> 11, t = mg & (SEQ - 1);
          int h = col >> 6, d = col & (HD - 1);
          Vtb[(((size_t)bi * NH + h) * HD + d) * SEQ + t] = (bf16_t)acc[f * 4 + nb][j];
        }
  } else {
    bf16_t* dst = (z == 0) ? Qb : Kb;
    #pragma unroll
    for (int f = 0; f < 2; ++f)
      #pragma unroll
      for (int nb = 0; nb < 4; ++nb)
        #pragma unroll
        for (int j = 0; j < 8; ++j) {
          int mg  = m0 + wave * 32 + f * 16 + j + 8 * half;
          int col = n0 + nb * 16 + row;
          int bi = mg >> 11, t = mg & (SEQ - 1);
          int h = col >> 6, d = col & (HD - 1);
          dst[(((size_t)bi * NH + h) * SEQ + t) * HD + d] = (bf16_t)acc[f * 4 + nb][j];
        }
  }
}

// ---------------------------------------------------------------------------
// Kernel 2: causal flash attention. grid = (SEQ/16, NH, BATCH), block = 32.
// One wave owns 16 query rows. Per 32-key step: 4 bf16 WMMAs build two
// 16x16 score tiles (B = K^T loads contiguously from K rows), streaming
// softmax with 16-lane shfl reductions, P repacked C->A layout via LDS,
// then 4 WMMAs accumulate P @ V against transposed V.
// Quirk preserved: scale = 1/sqrt(H) = 0.25.
// ---------------------------------------------------------------------------
__global__ __launch_bounds__(32)
void attn_kernel(const bf16_t* __restrict__ Qb,
                 const bf16_t* __restrict__ Kb,
                 const bf16_t* __restrict__ Vtb,
                 bf16_t* __restrict__ ctx) {
  __shared__ alignas(64) bf16_t Pl[16][32];

  const int lane = threadIdx.x;
  const int row  = lane & 15;
  const int half = lane >> 4;
  const int q0 = blockIdx.x * 16;
  const int h  = blockIdx.y;
  const int bi = blockIdx.z;

  const bf16_t* Qp  = Qb  + ((size_t)bi * NH + h) * SEQ * HD;
  const bf16_t* Kp  = Kb  + ((size_t)bi * NH + h) * SEQ * HD;
  const bf16_t* Vtp = Vtb + ((size_t)bi * NH + h) * HD * SEQ;

  const bf16_t* qrow = Qp + (size_t)(q0 + row) * HD;
  v16bf aq0 = cat8(*(const v8bf*)(qrow +      8 * half),
                   *(const v8bf*)(qrow + 16 + 8 * half));
  v16bf aq1 = cat8(*(const v8bf*)(qrow + 32 + 8 * half),
                   *(const v8bf*)(qrow + 48 + 8 * half));

  const v8f vzero = {0.f,0.f,0.f,0.f,0.f,0.f,0.f,0.f};
  float rm[8], rl[8];
  v8f acc[4];
  #pragma unroll
  for (int nb = 0; nb < 4; ++nb) acc[nb] = vzero;
  #pragma unroll
  for (int j = 0; j < 8; ++j) { rm[j] = -1e30f; rl[j] = 0.0f; }

  const int niter = (q0 >> 5) + 1;
  for (int it = 0; it < niter; ++it) {
    const int k0 = it * 32;
    const bf16_t* kr0 = Kp + (size_t)(k0 + row) * HD;
    const bf16_t* kr1 = Kp + (size_t)(k0 + 16 + row) * HD;

    v8f s0 = vzero, s1 = vzero;
    s0 = WMMA_BF16(aq0, cat8(*(const v8bf*)(kr0 +      8 * half), *(const v8bf*)(kr0 + 16 + 8 * half)), s0);
    s0 = WMMA_BF16(aq1, cat8(*(const v8bf*)(kr0 + 32 + 8 * half), *(const v8bf*)(kr0 + 48 + 8 * half)), s0);
    s1 = WMMA_BF16(aq0, cat8(*(const v8bf*)(kr1 +      8 * half), *(const v8bf*)(kr1 + 16 + 8 * half)), s1);
    s1 = WMMA_BF16(aq1, cat8(*(const v8bf*)(kr1 + 32 + 8 * half), *(const v8bf*)(kr1 + 48 + 8 * half)), s1);

    #pragma unroll
    for (int j = 0; j < 8; ++j) {
      const int q = q0 + j + 8 * half;
      float v0 = s0[j] * 0.25f; if (k0 + row      > q) v0 = -1e30f;
      float v1 = s1[j] * 0.25f; if (k0 + 16 + row > q) v1 = -1e30f;
      float tmax = fmaxf(v0, v1);
      #pragma unroll
      for (int off = 8; off >= 1; off >>= 1)
        tmax = fmaxf(tmax, __shfl_xor(tmax, off, 16));
      const float mnew = fmaxf(rm[j], tmax);
      const float p0 = __expf(v0 - mnew);
      const float p1 = __expf(v1 - mnew);
      const float f  = __expf(rm[j] - mnew);
      float ps = p0 + p1;
      #pragma unroll
      for (int off = 8; off >= 1; off >>= 1)
        ps += __shfl_xor(ps, off, 16);
      rl[j] = rl[j] * f + ps;
      rm[j] = mnew;
      #pragma unroll
      for (int nb = 0; nb < 4; ++nb) acc[nb][j] *= f;
      Pl[j + 8 * half][row]      = (bf16_t)p0;
      Pl[j + 8 * half][16 + row] = (bf16_t)p1;
    }
    __syncthreads();
    v16bf ap = cat8(*(const v8bf*)&Pl[row][8 * half],
                    *(const v8bf*)&Pl[row][16 + 8 * half]);
    #pragma unroll
    for (int nb = 0; nb < 4; ++nb) {
      const bf16_t* vr = Vtp + (size_t)(nb * 16 + row) * SEQ + k0 + 8 * half;
      acc[nb] = WMMA_BF16(ap, cat8(*(const v8bf*)vr, *(const v8bf*)(vr + 16)), acc[nb]);
    }
    __syncthreads();
  }

  #pragma unroll
  for (int j = 0; j < 8; ++j) {
    const float inv = 1.0f / rl[j];
    const int q = q0 + j + 8 * half;
    size_t base = ((size_t)bi * SEQ + q) * DOUT + h * HD;
    #pragma unroll
    for (int nb = 0; nb < 4; ++nb)
      ctx[base + nb * 16 + row] = (bf16_t)(acc[nb][j] * inv);
  }
}

// ---------------------------------------------------------------------------
// Kernel 3: output projection ctx @ Wo + bo -> f32.
// grid = (MTOT/128, DOUT/64), block = 128. Both A (bf16 ctx) and B
// (pre-transposed bf16 Wo) tiles staged via async global->LDS DMA.
// ---------------------------------------------------------------------------
__global__ __launch_bounds__(128)
void oproj_gemm_kernel(const bf16_t* __restrict__ ctx,
                       const bf16_t* __restrict__ Wot,  // [DOUT][DOUT] bf16
                       const float* __restrict__ bo,
                       float* __restrict__ out) {
  __shared__ alignas(64) bf16_t As[128][32];
  __shared__ alignas(64) bf16_t Ws[64][32];

  const int m0 = blockIdx.x * 128;
  const int n0 = blockIdx.y * 64;
  const int tid  = threadIdx.x;
  const int wave = tid >> 5;
  const int lane = tid & 31;
  const int row  = lane & 15;
  const int half = lane >> 4;

  const v8f vzero = {0.f,0.f,0.f,0.f,0.f,0.f,0.f,0.f};
  v8f acc[8];
  #pragma unroll
  for (int i = 0; i < 8; ++i) acc[i] = vzero;

  for (int k0 = 0; k0 < DOUT; k0 += 32) {
    __syncthreads();
    #pragma unroll
    for (int i = 0; i < 4; ++i) {
      int idx = i * 1024 + tid * 8;
      int r = idx >> 5, c = idx & 31;
      cp16_g2l(ctx + (size_t)(m0 + r) * DOUT + k0 + c, &As[r][c]);
    }
    #pragma unroll
    for (int i = 0; i < 2; ++i) {
      int idx = i * 1024 + tid * 8;
      int r = idx >> 5, c = idx & 31;
      cp16_g2l(Wot + (size_t)(n0 + r) * DOUT + k0 + c, &Ws[r][c]);
    }
    async_wait0();
    __syncthreads();

    v16bf a0 = cat8(*(const v8bf*)&As[wave * 32 + row][8 * half],
                    *(const v8bf*)&As[wave * 32 + row][16 + 8 * half]);
    v16bf a1 = cat8(*(const v8bf*)&As[wave * 32 + 16 + row][8 * half],
                    *(const v8bf*)&As[wave * 32 + 16 + row][16 + 8 * half]);
    #pragma unroll
    for (int nb = 0; nb < 4; ++nb) {
      v16bf b = cat8(*(const v8bf*)&Ws[nb * 16 + row][8 * half], *(const v8bf*)&Ws[nb * 16 + row][16 + 8 * half]);
      acc[nb]     = WMMA_BF16(a0, b, acc[nb]);
      acc[4 + nb] = WMMA_BF16(a1, b, acc[4 + nb]);
    }
  }

  #pragma unroll
  for (int f = 0; f < 2; ++f)
    #pragma unroll
    for (int nb = 0; nb < 4; ++nb)
      #pragma unroll
      for (int j = 0; j < 8; ++j) {
        int mg  = m0 + wave * 32 + f * 16 + j + 8 * half;
        int col = n0 + nb * 16 + row;
        out[(size_t)mg * DOUT + col] = acc[f * 4 + nb][j] + bo[col];
      }
}

// ---------------------------------------------------------------------------
extern "C" void kernel_launch(void* const* d_in, const int* in_sizes, int n_in,
                              void* d_out, int out_size, void* d_ws, size_t ws_size,
                              hipStream_t stream) {
  (void)in_sizes; (void)n_in; (void)out_size; (void)ws_size;
  const float* x  = (const float*)d_in[0];
  const float* Wq = (const float*)d_in[1];
  const float* Wk = (const float*)d_in[2];
  const float* Wv = (const float*)d_in[3];
  const float* Wo = (const float*)d_in[4];
  const float* bo = (const float*)d_in[5];
  float* out = (float*)d_out;

  char* ws = (char*)d_ws;
  const size_t SEG = (size_t)BATCH * NH * SEQ * HD * sizeof(bf16_t); // 16 MB
  bf16_t* Qb  = (bf16_t*)(ws + 0 * SEG);
  bf16_t* Kb  = (bf16_t*)(ws + 1 * SEG);
  bf16_t* Vtb = (bf16_t*)(ws + 2 * SEG);
  bf16_t* ctx = (bf16_t*)(ws + 3 * SEG);
  bf16_t* Wt  = (bf16_t*)(ws + 4 * SEG);   // 4 x [1024][1024] bf16 = 8 MB

  dim3 gw(DOUT / 32, DIN / 32, 4);
  wtrans_kernel<<<gw, dim3(32, 8), 0, stream>>>(Wq, Wk, Wv, Wo, Wt);

  dim3 gq(MTOT / 128, DOUT / 64, 3);
  qkv_gemm_kernel<<<gq, 128, 0, stream>>>(x, Wt, Qb, Kb, Vtb);

  dim3 ga(SEQ / 16, NH, BATCH);
  attn_kernel<<<ga, 32, 0, stream>>>(Qb, Kb, Vtb, ctx);

  dim3 go(MTOT / 128, DOUT / 64);
  oproj_gemm_kernel<<<go, 128, 0, stream>>>(ctx, Wt + 3 * (size_t)DIN * DOUT, bo, out);
}
